// RelationalLatentDynamics_76312978915364
// MI455X (gfx1250) — matrix operand order repaired and checked by reading, not compiled, not run on hardware
//
#include <hip/hip_runtime.h>
#include <hip/hip_bf16.h>

typedef __attribute__((ext_vector_type(16))) _Float16 v16h;
typedef __attribute__((ext_vector_type(8)))  _Float16 h8;
typedef __attribute__((ext_vector_type(8)))  float    v8f;

// ---------------- workspace layout: f16 weights, transposed [N][K], K padded ----
constexpr int OFF_EW1A = 0;               // 64 x 128
constexpr int OFF_EW1B = OFF_EW1A + 64*128;
constexpr int OFF_EW2  = OFF_EW1B + 64*128;   // 64 x 64
constexpr int OFF_EW3  = OFF_EW2  + 64*64;    // 64 x 64
constexpr int OFF_N1   = OFF_EW3  + 64*64;    // 64 x 192
constexpr int OFF_N2   = OFF_N1   + 64*192;   // 64 x 64
constexpr int OFF_N3   = OFF_N2   + 64*64;    // 32 x 64
constexpr int OFF_N4   = OFF_N3   + 32*64;    // 32 x 32
constexpr int OFF_L1A  = OFF_N4   + 32*32;    // 64 x 64 (K padded 48->64)
constexpr int OFF_L1B  = OFF_L1A  + 64*64;
constexpr int OFF_L2   = OFF_L1B  + 64*64;
constexpr int OFF_L3   = OFF_L2   + 64*64;
constexpr int OFF_I1   = OFF_L3   + 64*64;    // 64 x 128 (K padded 112->128)
constexpr int OFF_I2   = OFF_I1   + 64*128;
constexpr int OFF_I3   = OFF_I2   + 64*64;    // 32 x 64
constexpr int OFF_I4   = OFF_I3   + 32*64;    // 32 x 32
constexpr int OFF_ET1  = OFF_I4   + 32*32;    // 32 x 32
constexpr int OFF_ET2  = OFF_ET1  + 32*32;    // 64 x 32
constexpr int OFF_ET3  = OFF_ET2  + 64*32;    // 32 x 64
constexpr int OFF_ET4  = OFF_ET3  + 32*64;    // 16 x 32
constexpr int WS_HALFS = OFF_ET4  + 16*32;

struct Args { const void* p[40]; };

// prep jobs: {d_in idx, din(src rows), dout, padK, srcRowOff, dstOff}
__device__ const int g_jobs[20][6] = {
    { 1, 128, 64, 128,   0, OFF_EW1A},
    { 1, 128, 64, 128, 128, OFF_EW1B},
    { 3,  64, 64,  64,   0, OFF_EW2},
    { 5,  64, 64,  64,   0, OFF_EW3},
    { 7, 192, 64, 192,   0, OFF_N1},
    { 9,  64, 64,  64,   0, OFF_N2},
    {11,  64, 32,  64,   0, OFF_N3},
    {13,  32, 32,  32,   0, OFF_N4},
    {15,  48, 64,  64,   0, OFF_L1A},
    {15,  48, 64,  64,  48, OFF_L1B},
    {17,  64, 64,  64,   0, OFF_L2},
    {19,  64, 64,  64,   0, OFF_L3},
    {21, 112, 64, 128,   0, OFF_I1},
    {23,  64, 64,  64,   0, OFF_I2},
    {25,  64, 32,  64,   0, OFF_I3},
    {27,  32, 32,  32,   0, OFF_I4},
    {31,  32, 32,  32,   0, OFF_ET1},
    {33,  32, 64,  32,   0, OFF_ET2},
    {35,  64, 32,  64,   0, OFF_ET3},
    {37,  32, 16,  32,   0, OFF_ET4},
};

// Convert f32 weights (din,dout) -> f16 transposed [dout][padK], zero K-pad.
__global__ __launch_bounds__(256) void rld_prep(Args a, _Float16* __restrict__ W) {
    int j = blockIdx.x;
    int ptrIdx = g_jobs[j][0], din = g_jobs[j][1], dout = g_jobs[j][2];
    int padK = g_jobs[j][3], roff = g_jobs[j][4], off = g_jobs[j][5];
    const float* w = (const float*)a.p[ptrIdx];
    int total = dout * padK;
    for (int idx = threadIdx.x; idx < total; idx += 256) {
        int n = idx / padK, k = idx % padK;
        float v = (k < din) ? w[(roff + k) * dout + n] : 0.0f;
        W[off + n * padK + k] = (_Float16)v;
    }
}

// ---------------- WMMA fragment loaders (CDNA5 16x16x32 f16 layouts) ----------
__device__ __forceinline__ v16h load_a_frag(const _Float16* A, int lda, int mt,
                                            int k0, int lane) {
    // A 16x32 f16: lanes 0-15 -> M=lane, K base 0; lanes 16-31 -> M=lane-16, K base 8.
    // per lane: elems 0..7 = K kb..kb+7, elems 8..15 = K kb+16..kb+23.
    int m  = mt * 16 + (lane & 15);
    int kb = (lane >> 4) << 3;
    const _Float16* ap = A + m * lda + k0 + kb;
    h8 lo = *(const h8*)(ap);
    h8 hi = *(const h8*)(ap + 16);
    v16h r;
#pragma unroll
    for (int e = 0; e < 8; ++e) { r[e] = lo[e]; r[e + 8] = hi[e]; }
    return r;
}

__device__ __forceinline__ v16h load_b_frag(const _Float16* Bt, int ldb, int nt,
                                            int k0, int lane) {
    // B 32x16 f16, stored transposed [N][K]: lane n = col; lanes 0-15 K base 0,
    // lanes 16-31 K base 8; elems 0..7 = K kb..kb+7, elems 8..15 = K kb+16..kb+23.
    int n  = nt * 16 + (lane & 15);
    int kb = (lane >> 4) << 3;
    const _Float16* bp = Bt + n * ldb + k0 + kb;
    h8 lo = *(const h8*)(bp);
    h8 hi = *(const h8*)(bp + 16);
    v16h r;
#pragma unroll
    for (int e = 0; e < 8; ++e) { r[e] = lo[e]; r[e + 8] = hi[e]; }
    return r;
}

// C = A(MxK,f16,LDS) @ B(KxN,f16 transposed,global) [+bias][relu] -> f16 LDS out
__device__ void gemm_tile(const _Float16* A, int lda,
                          const _Float16* Bt, int ldb,
                          _Float16* Out, int ldo,
                          const float* bias,
                          int M, int N, int K, bool relu,
                          int wave, int lane) {
    int ntn = N >> 4;
    int ntiles = (M >> 4) * ntn;
    for (int t = wave; t < ntiles; t += 8) {
        int mt = t / ntn, nt = t % ntn;
        v8f acc = {};
        for (int k0 = 0; k0 < K; k0 += 32) {
            v16h af = load_a_frag(A, lda, mt, k0, lane);
            v16h bf = load_b_frag(Bt, ldb, nt, k0, lane);
            acc = __builtin_amdgcn_wmma_f32_16x16x32_f16(
                false, af, false, bf, (short)0, acc, false, false);
        }
        int n = nt * 16 + (lane & 15);
        float bv = bias ? bias[n] : 0.0f;
        int mbase = mt * 16 + ((lane >> 4) << 3);
#pragma unroll
        for (int r = 0; r < 8; ++r) {
            float v = acc[r] + bv;
            if (relu) v = v > 0.0f ? v : 0.0f;
            Out[(mbase + r) * ldo + n] = (_Float16)v;
        }
    }
}

// Edge output layer: e = H2 @ W3 + b3; agg[i][n] += e over pairs j != i (ds_add_f32)
__device__ void gemm_edge(const _Float16* A, int lda,
                          const _Float16* Bt, int ldb,
                          float* agg, const float* bias, int ibase,
                          int Mchunk, int N, int K,
                          int wave, int lane) {
    int ntn = N >> 4;
    int ntiles = (Mchunk >> 4) * ntn;
    for (int t = wave; t < ntiles; t += 8) {
        int mt = t / ntn, nt = t % ntn;
        v8f acc = {};
        for (int k0 = 0; k0 < K; k0 += 32) {
            v16h af = load_a_frag(A, lda, mt, k0, lane);
            v16h bf = load_b_frag(Bt, ldb, nt, k0, lane);
            acc = __builtin_amdgcn_wmma_f32_16x16x32_f16(
                false, af, false, bf, (short)0, acc, false, false);
        }
        int n = nt * 16 + (lane & 15);
        float bv = bias[n];
        int mbase = mt * 16 + ((lane >> 4) << 3);
#pragma unroll
        for (int r = 0; r < 8; ++r) {
            int p = mbase + r;          // pair row within chunk
            int i = ibase + (p >> 5);   // target object
            int j = p & 31;             // neighbor
            if (j != i) atomicAdd(&agg[i * 64 + n], acc[r] + bv);
        }
    }
}

// ---------------- main kernel: one workgroup per batch element ----------------
__global__ __launch_bounds__(256) void rld_main(Args a, const _Float16* __restrict__ W,
                                                float* __restrict__ out) {
    __shared__ _Float16 sSrc[32 * 192];  // src(128) | agg(64); reused as x in rollout
    __shared__ _Float16 sU[32 * 64];
    __shared__ _Float16 sV[32 * 64];
    __shared__ _Float16 sH1[64 * 64];
    __shared__ _Float16 sH2[64 * 64];
    __shared__ float    sAgg[32 * 64];
    __shared__ _Float16 nA[32 * 64];
    __shared__ _Float16 nB[32 * 64];
    __shared__ _Float16 sZu[32 * 64];
    __shared__ float    sZ[32 * 48];     // [expl(16) | impl(32)] per object

    const int b    = blockIdx.x;
    const int tid  = threadIdx.x;
    const int wave = tid >> 5;
    const int lane = tid & 31;

    const float* z = (const float*)a.p[0];
    if (tid < 8) __builtin_prefetch(W + tid * (WS_HALFS / 8), 0, 1);

    // ---- build source (B,O,T*E) and init explicit state from last timestep ----
    for (int idx = tid; idx < 32 * 128; idx += 256) {
        int o = idx >> 7, c = idx & 127;
        int t = c >> 4, e = c & 15;
        sSrc[o * 192 + c] = (_Float16)z[(((size_t)b * 8 + t) * 32 + o) * 16 + e];
    }
    for (int idx = tid; idx < 32 * 16; idx += 256) {
        int o = idx >> 4, e = idx & 15;
        sZ[o * 48 + e] = z[(((size_t)b * 8 + 7) * 32 + o) * 16 + e];
    }
    __syncthreads();

    // ---- encoder edges: factorized first layer U = s@W1a+b0, V = s@W1b ----
    gemm_tile(sSrc, 192, W + OFF_EW1A, 128, sU, 64, (const float*)a.p[2],
              32, 64, 128, false, wave, lane);
    gemm_tile(sSrc, 192, W + OFF_EW1B, 128, sV, 64, nullptr,
              32, 64, 128, false, wave, lane);
    __syncthreads();
    for (int idx = tid; idx < 32 * 64; idx += 256) sAgg[idx] = 0.0f;
    __syncthreads();
    for (int ig = 0; ig < 16; ++ig) {            // 2 objects x 32 neighbors per chunk
        for (int idx = tid; idx < 64 * 64; idx += 256) {
            int p = idx >> 6, n = idx & 63;
            int i = ig * 2 + (p >> 5), j = p & 31;
            float v = (float)sU[i * 64 + n] + (float)sV[j * 64 + n];
            sH1[p * 64 + n] = (_Float16)(v > 0.0f ? v : 0.0f);
        }
        __syncthreads();
        gemm_tile(sH1, 64, W + OFF_EW2, 64, sH2, 64, (const float*)a.p[4],
                  64, 64, 64, true, wave, lane);
        __syncthreads();
        gemm_edge(sH2, 64, W + OFF_EW3, 64, sAgg, (const float*)a.p[6],
                  ig * 2, 64, 64, 64, wave, lane);
        __syncthreads();
    }
    for (int idx = tid; idx < 32 * 64; idx += 256) {
        int o = idx >> 6, n = idx & 63;
        sSrc[o * 192 + 128 + n] = (_Float16)sAgg[idx];
    }
    __syncthreads();

    // ---- node MLP [192->64->64->32->32] -> z_impl ----
    gemm_tile(sSrc, 192, W + OFF_N1, 192, nA, 64, (const float*)a.p[8],  32, 64, 192, true,  wave, lane); __syncthreads();
    gemm_tile(nA,   64,  W + OFF_N2, 64,  nB, 64, (const float*)a.p[10], 32, 64, 64,  true,  wave, lane); __syncthreads();
    gemm_tile(nB,   64,  W + OFF_N3, 64,  nA, 64, (const float*)a.p[12], 32, 32, 64,  true,  wave, lane); __syncthreads();
    gemm_tile(nA,   64,  W + OFF_N4, 32,  nB, 64, (const float*)a.p[14], 32, 32, 32,  false, wave, lane); __syncthreads();
    for (int idx = tid; idx < 32 * 32; idx += 256) {
        int o = idx >> 5, c = idx & 31;
        sZ[o * 48 + 16 + c] = (float)nB[o * 64 + c];
    }
    __syncthreads();

    // ---- rollout ----
    int tf = *(const int*)a.p[39];
    if (tf > 8) tf = 8;
    for (int t = 0; t < tf; ++t) {
        // et-MLP [32->32->64->32->16] on z_impl
        for (int idx = tid; idx < 32 * 32; idx += 256) {
            int o = idx >> 5, c = idx & 31;
            nA[o * 64 + c] = (_Float16)sZ[o * 48 + 16 + c];
        }
        __syncthreads();
        gemm_tile(nA, 64, W + OFF_ET1, 32, nB, 64, (const float*)a.p[32], 32, 32, 32, true,  wave, lane); __syncthreads();
        gemm_tile(nB, 64, W + OFF_ET2, 32, nA, 64, (const float*)a.p[34], 32, 64, 32, true,  wave, lane); __syncthreads();
        gemm_tile(nA, 64, W + OFF_ET3, 64, nB, 64, (const float*)a.p[36], 32, 32, 64, true,  wave, lane); __syncthreads();
        gemm_tile(nB, 64, W + OFF_ET4, 32, nA, 64, (const float*)a.p[38], 32, 16, 32, false, wave, lane); __syncthreads();
        // z_expl += dE ; emit prediction for this step
        for (int idx = tid; idx < 32 * 16; idx += 256) {
            int o = idx >> 4, e = idx & 15;
            float v = sZ[o * 48 + e] + (float)nA[o * 64 + e];
            sZ[o * 48 + e] = v;
            out[(((size_t)b * 8 + t) * 32 + o) * 16 + e] = v;
        }
        __syncthreads();

        // ledge edges (factorized, K padded 48->64)
        for (int idx = tid; idx < 32 * 64; idx += 256) {
            int o = idx >> 6, c = idx & 63;
            sZu[idx] = (_Float16)((c < 48) ? sZ[o * 48 + c] : 0.0f);
        }
        __syncthreads();
        gemm_tile(sZu, 64, W + OFF_L1A, 64, sU, 64, (const float*)a.p[16], 32, 64, 64, false, wave, lane);
        gemm_tile(sZu, 64, W + OFF_L1B, 64, sV, 64, nullptr,               32, 64, 64, false, wave, lane);
        __syncthreads();
        for (int idx = tid; idx < 32 * 64; idx += 256) sAgg[idx] = 0.0f;
        __syncthreads();
        for (int ig = 0; ig < 16; ++ig) {
            for (int idx = tid; idx < 64 * 64; idx += 256) {
                int p = idx >> 6, n = idx & 63;
                int i = ig * 2 + (p >> 5), j = p & 31;
                float v = (float)sU[i * 64 + n] + (float)sV[j * 64 + n];
                sH1[p * 64 + n] = (_Float16)(v > 0.0f ? v : 0.0f);
            }
            __syncthreads();
            gemm_tile(sH1, 64, W + OFF_L2, 64, sH2, 64, (const float*)a.p[18],
                      64, 64, 64, true, wave, lane);
            __syncthreads();
            gemm_edge(sH2, 64, W + OFF_L3, 64, sAgg, (const float*)a.p[20],
                      ig * 2, 64, 64, 64, wave, lane);
            __syncthreads();
        }

        // x = [zu(48) | agg(64) | 0-pad(16)] (ld 192), it-MLP [128->64->64->32->32]
        for (int idx = tid; idx < 32 * 128; idx += 256) {
            int o = idx >> 7, c = idx & 127;
            float v = (c < 48) ? sZ[o * 48 + c]
                               : (c < 112 ? sAgg[o * 64 + (c - 48)] : 0.0f);
            sSrc[o * 192 + c] = (_Float16)v;
        }
        __syncthreads();
        gemm_tile(sSrc, 192, W + OFF_I1, 128, nA, 64, (const float*)a.p[22], 32, 64, 128, true,  wave, lane); __syncthreads();
        gemm_tile(nA,   64,  W + OFF_I2, 64,  nB, 64, (const float*)a.p[24], 32, 64, 64,  true,  wave, lane); __syncthreads();
        gemm_tile(nB,   64,  W + OFF_I3, 64,  nA, 64, (const float*)a.p[26], 32, 32, 64,  true,  wave, lane); __syncthreads();
        gemm_tile(nA,   64,  W + OFF_I4, 32,  nB, 64, (const float*)a.p[28], 32, 32, 32,  false, wave, lane); __syncthreads();

        // gate (112-dim dot, f32) + implicit-state update
        if (tid < 32) {
            int o = tid;
            const float* gw = (const float*)a.p[29];
            float acc = ((const float*)a.p[30])[0];
            for (int k = 0; k < 48; ++k) acc += sZ[o * 48 + k] * gw[k];
            for (int k = 0; k < 64; ++k) acc += sAgg[o * 64 + k] * gw[48 + k];
            float g = 1.0f / (1.0f + __expf(-acc));
            for (int c = 0; c < 32; ++c)
                sZ[o * 48 + 16 + c] += (float)nB[o * 64 + c] * g;
        }
        __syncthreads();
    }
}

extern "C" void kernel_launch(void* const* d_in, const int* in_sizes, int n_in,
                              void* d_out, int out_size, void* d_ws, size_t ws_size,
                              hipStream_t stream) {
    (void)in_sizes; (void)out_size; (void)ws_size;
    Args a{};
    for (int i = 0; i < 40 && i < n_in; ++i) a.p[i] = d_in[i];
    _Float16* W = (_Float16*)d_ws;
    rld_prep<<<20, 256, 0, stream>>>(a, W);
    rld_main<<<64, 256, 0, stream>>>(a, W, (float*)d_out);
}
